// EquivariantAttention_37177236914838
// MI455X (gfx1250) — matrix operand order repaired and checked by reading, not compiled, not run on hardware
//
#include <hip/hip_runtime.h>

// Problem constants (B=8, N=2048, H=512) from the reference.
#define B_  8
#define N_  2048
#define H_  512
#define H3_ 1536

typedef __bf16 bf16;
typedef bf16  v16bf __attribute__((ext_vector_type(16)));
typedef bf16  v8bf  __attribute__((ext_vector_type(8)));
typedef float v8f   __attribute__((ext_vector_type(8)));

static __device__ inline v8f zero8() {
  v8f z;
#pragma unroll
  for (int i = 0; i < 8; ++i) z[i] = 0.0f;
  return z;
}

static __device__ inline v16bf cat16(v8bf lo, v8bf hi) {
  v16bf r;
#pragma unroll
  for (int i = 0; i < 8; ++i) { r[i] = lo[i]; r[i + 8] = hi[i]; }
  return r;
}

// A-matrix fragment (16x32 bf16): lanes 0-15 hold K {0..15} of row (lane&15),
// lanes 16-31 hold K {16..31}. p points at &A[row*ld + kc].
static __device__ inline v16bf load_a_frag(const bf16* __restrict__ p, int hi) {
  v8bf lo = *(const v8bf*)(p + hi * 16);
  v8bf h2 = *(const v8bf*)(p + 8 + hi * 16);
  return cat16(lo, h2);
}

// B-matrix fragment (32x16 bf16): lane holds 16 contiguous K values of its column.
static __device__ inline v16bf load_b_frag(const bf16* __restrict__ p) {
  v8bf lo = *(const v8bf*)(p);
  v8bf h2 = *(const v8bf*)(p + 8);
  return cat16(lo, h2);
}

static __device__ inline v8f wmma_bf16(v16bf a, v16bf b, v8f c) {
  return __builtin_amdgcn_wmma_f32_16x16x32_bf16(false, a, false, b, (short)0, c,
                                                 false, false);
}

// xor-butterfly reductions across the 16 lanes of a wave half (C-layout rows).
static __device__ inline float half16_max(float v) {
  int lane = (int)(threadIdx.x & 31);
#pragma unroll
  for (int mask = 1; mask < 16; mask <<= 1) {
    int o = __builtin_amdgcn_ds_bpermute(((lane ^ mask) << 2), __float_as_int(v));
    v = fmaxf(v, __int_as_float(o));
  }
  return v;
}
static __device__ inline float half16_sum(float v) {
  int lane = (int)(threadIdx.x & 31);
#pragma unroll
  for (int mask = 1; mask < 16; mask <<= 1) {
    int o = __builtin_amdgcn_ds_bpermute(((lane ^ mask) << 2), __float_as_int(v));
    v = v + __int_as_float(o);
  }
  return v;
}

// ---------------- pre-pass kernels ----------------

__global__ void k_f32_to_bf16(const float* __restrict__ in, bf16* __restrict__ out,
                              int n) {
  int i = blockIdx.x * blockDim.x + threadIdx.x;
  if (i < n) out[i] = (bf16)in[i];
}

// W[K][Nn] (row-major f32) -> Wt[Nn][K] (row-major bf16)
__global__ void k_transpose_w(const float* __restrict__ W, bf16* __restrict__ Wt,
                              int K, int Nn) {
  int i = blockIdx.x * blockDim.x + threadIdx.x;
  if (i < K * Nn) {
    int n = i / K, k = i % K;
    Wt[(size_t)n * K + k] = (bf16)W[(size_t)k * Nn + n];
  }
}

// ---------------- QKV projection GEMM ----------------
// C(16384x1536) = Xb @ Wqkv + b. Block: 256 thr (8 waves) -> 128x128 block tile,
// wave tile 32x64 (8 accums). Q,K stored row-major bf16; V stored TRANSPOSED
// per batch as Vt[batch][h][n] so attention B-frags are contiguous.
__global__ __launch_bounds__(256, 1)
void k_qkv_gemm(const bf16* __restrict__ Xb, const bf16* __restrict__ Wt,
                const float* __restrict__ bias, bf16* __restrict__ Qb,
                bf16* __restrict__ Kb, bf16* __restrict__ Vtg) {
  const int lane = threadIdx.x & 31;
  const int wave = threadIdx.x >> 5;
  const int m  = lane & 15;
  const int hi = lane >> 4;
  const int rw = wave & 3;                    // 4 row groups of 32
  const int cw = wave >> 2;                   // 2 col groups of 64
  const int rbase = blockIdx.x * 128 + rw * 32;
  const int n0    = blockIdx.y * 128 + cw * 64;

  const bf16* arow0 = Xb + (size_t)(rbase + m) * H_;
  const bf16* arow1 = arow0 + (size_t)16 * H_;

  v8f acc[2][4];
#pragma unroll
  for (int r = 0; r < 2; ++r)
#pragma unroll
    for (int t = 0; t < 4; ++t) acc[r][t] = zero8();

  // software pipeline: frags for k-chunk kc live while kc+32 is being fetched
  v16bf a0 = load_a_frag(arow0, hi);
  v16bf a1 = load_a_frag(arow1, hi);
  v16bf bfr[4];
#pragma unroll
  for (int t = 0; t < 4; ++t)
    bfr[t] = load_b_frag(Wt + (size_t)(n0 + t * 16 + m) * H_ + hi * 16);

#pragma unroll
  for (int kc = 0; kc < H_; kc += 32) {
    v16bf na0, na1, nb[4];
    if (kc + 32 < H_) {
      na0 = load_a_frag(arow0 + kc + 32, hi);
      na1 = load_a_frag(arow1 + kc + 32, hi);
#pragma unroll
      for (int t = 0; t < 4; ++t)
        nb[t] = load_b_frag(Wt + (size_t)(n0 + t * 16 + m) * H_ + kc + 32 + hi * 16);
    }
#pragma unroll
    for (int t = 0; t < 4; ++t) {
      acc[0][t] = wmma_bf16(a0, bfr[t], acc[0][t]);
      acc[1][t] = wmma_bf16(a1, bfr[t], acc[1][t]);
    }
    if (kc + 32 < H_) {
      a0 = na0; a1 = na1;
#pragma unroll
      for (int t = 0; t < 4; ++t) bfr[t] = nb[t];
    }
  }

#pragma unroll
  for (int r = 0; r < 2; ++r) {
#pragma unroll
    for (int t = 0; t < 4; ++t) {
      int cg = n0 + t * 16 + m;               // 0..1535
      float bv = bias[cg];
      int sec = cg >> 9;                      // 0=Q 1=K 2=V
      int cin = cg & (H_ - 1);
      if (sec < 2) {
        bf16* dst = (sec == 0) ? Qb : Kb;
#pragma unroll
        for (int e = 0; e < 8; ++e) {
          int row = rbase + r * 16 + e + 8 * hi;   // C-layout M = e + 8*hi
          dst[(size_t)row * H_ + cin] = (bf16)(acc[r][t][e] + bv);
        }
      } else {
        // V transposed: 8 consecutive n-values in one lane -> packed b128 store
        int row0 = rbase + r * 16 + 8 * hi;
        int batch = row0 >> 11;               // /N_
        int nn = row0 & (N_ - 1);
        v8bf pk;
#pragma unroll
        for (int e = 0; e < 8; ++e) pk[e] = (bf16)(acc[r][t][e] + bv);
        *(v8bf*)(Vtg + (size_t)batch * H_ * N_ + (size_t)cin * N_ + nn) = pk;
      }
    }
  }
}

// ---------------- flash attention ----------------
// Block: 128 threads (4 waves), wave owns 16 Q rows; keys processed 64 at a
// time. K and V^T B-frags load contiguously straight from global; only the
// softmax P tile is relayouted through a 2 KB per-wave LDS patch.
__global__ __launch_bounds__(128, 1)
void k_attn(const bf16* __restrict__ Qb, const bf16* __restrict__ Kb,
            const bf16* __restrict__ Vtg, const float* __restrict__ geo,
            bf16* __restrict__ Ob) {
  __shared__ bf16 Pst[4][16 * 64];            // per-wave P staging, 8 KB total

  const int lane = threadIdx.x & 31;
  const int wave = threadIdx.x >> 5;
  const int c  = lane & 15;
  const int hi = lane >> 4;
  const int nb = N_ / 64;
  const int batch = blockIdx.x / nb;
  const int qrow0 = (blockIdx.x % nb) * 64 + wave * 16;
  const size_t rowbase = (size_t)batch * N_;
  const bf16* vbatch = Vtg + (size_t)batch * H_ * N_;
  const float scale = 0.04419417382415922f;   // 1/sqrt(512)

  // Resident Q fragments: 16 rows x 512 (16 K-chunks), 128 VGPRs.
  v16bf qf[16];
  const bf16* qrow = Qb + (rowbase + qrow0 + c) * H_;
#pragma unroll
  for (int kc = 0; kc < 16; ++kc) qf[kc] = load_a_frag(qrow + kc * 32, hi);

  // Geometry of my 8 C-layout rows (row = e + 8*hi), replicated across half.
  float gix[8], giy[8], giz[8];
#pragma unroll
  for (int e = 0; e < 8; ++e) {
    const float* g = geo + (rowbase + qrow0 + e + 8 * hi) * 3;
    gix[e] = g[0]; giy[e] = g[1]; giz[e] = g[2];
  }

  float mrow[8], lrow[8];
#pragma unroll
  for (int e = 0; e < 8; ++e) { mrow[e] = -1e30f; lrow[e] = 0.0f; }

  v8f Oacc[32];                               // 16 rows x 512 f32 = 256 VGPRs
#pragma unroll
  for (int t = 0; t < 32; ++t) Oacc[t] = zero8();

#pragma unroll 1
  for (int jb = 0; jb < N_; jb += 64) {
    // ---- S = Q @ K^T for 64 keys: four 16x16 accumulators ----
    v8f S[4];
#pragma unroll
    for (int q = 0; q < 4; ++q) S[q] = zero8();
#pragma unroll
    for (int kc = 0; kc < 16; ++kc) {
#pragma unroll
      for (int q = 0; q < 4; ++q) {
        const bf16* kp = Kb + (rowbase + jb + q * 16 + c) * H_ + kc * 32 + hi * 16;
        S[q] = wmma_bf16(qf[kc], load_b_frag(kp), S[q]);
      }
    }

    // ---- geometric decay + online softmax ----
    float gjx[4], gjy[4], gjz[4];
#pragma unroll
    for (int q = 0; q < 4; ++q) {
      const float* g = geo + (rowbase + jb + q * 16 + c) * 3;
      gjx[q] = g[0]; gjy[q] = g[1]; gjz[q] = g[2];
    }

    float oscale[8];
    bf16* pw = &Pst[wave][0];
#pragma unroll
    for (int e = 0; e < 8; ++e) {
      float s[4];
#pragma unroll
      for (int q = 0; q < 4; ++q) {
        float dx = gix[e] - gjx[q], dy = giy[e] - gjy[q], dz = giz[e] - gjz[q];
        float d = sqrtf(dx * dx + dy * dy + dz * dz);
        s[q] = S[q][e] * scale * __expf(-d);
      }
      float mx = fmaxf(fmaxf(s[0], s[1]), fmaxf(s[2], s[3]));
      mx = half16_max(mx);
      float mnew = fmaxf(mrow[e], mx);
      float p[4], psum = 0.0f;
#pragma unroll
      for (int q = 0; q < 4; ++q) { p[q] = __expf(s[q] - mnew); psum += p[q]; }
      float rs = half16_sum(psum);
      float sc = __expf(mrow[e] - mnew);
      lrow[e] = lrow[e] * sc + rs;
      mrow[e] = mnew;
      oscale[e] = sc;
#pragma unroll
      for (int q = 0; q < 4; ++q)
        pw[(e + 8 * hi) * 64 + q * 16 + c] = (bf16)p[q];   // C-layout row
    }

    // Rescale running O.
#pragma unroll
    for (int t = 0; t < 32; ++t)
#pragma unroll
      for (int e = 0; e < 8; ++e) Oacc[t][e] *= oscale[e];

    __syncthreads();   // order P stores before cross-lane P frag reads

    // Reload P as two A-fragments (keys 0..31 and 32..63).
    const bf16* pr = &Pst[wave][0];
    v16bf pf0 = cat16(*(const v8bf*)(pr + c * 64 + hi * 16),
                      *(const v8bf*)(pr + c * 64 + 8 + hi * 16));
    v16bf pf1 = cat16(*(const v8bf*)(pr + c * 64 + 32 + hi * 16),
                      *(const v8bf*)(pr + c * 64 + 40 + hi * 16));

    // ---- O += P @ V: V^T B-frags contiguous straight from global ----
#pragma unroll
    for (int t = 0; t < 32; ++t) {
      const bf16* vp0 = vbatch + (size_t)(t * 16 + c) * N_ + jb + hi * 16;
      v8f o = wmma_bf16(pf0, load_b_frag(vp0), Oacc[t]);
      Oacc[t] = wmma_bf16(pf1, load_b_frag(vp0 + 32), o);
    }
  }

  // Normalize and emit bf16 O for the output projection.
  float inv_l[8];
#pragma unroll
  for (int e = 0; e < 8; ++e) inv_l[e] = 1.0f / lrow[e];
#pragma unroll
  for (int t = 0; t < 32; ++t) {
    int h = t * 16 + c;
#pragma unroll
    for (int e = 0; e < 8; ++e) {
      size_t row = rowbase + qrow0 + e + 8 * hi;
      Ob[row * H_ + h] = (bf16)(Oacc[t][e] * inv_l[e]);
    }
  }
}

// ---------------- output projection ----------------
// out(16384x512) = Ob @ Wout + b_out, f32 result. Same 32x64 wave tiling.
__global__ __launch_bounds__(256, 1)
void k_out_gemm(const bf16* __restrict__ Ab, const bf16* __restrict__ Wt,
                const float* __restrict__ bias, float* __restrict__ out) {
  const int lane = threadIdx.x & 31;
  const int wave = threadIdx.x >> 5;
  const int m  = lane & 15;
  const int hi = lane >> 4;
  const int rw = wave & 3;
  const int cw = wave >> 2;
  const int rbase = blockIdx.x * 128 + rw * 32;
  const int n0    = blockIdx.y * 128 + cw * 64;

  const bf16* arow0 = Ab + (size_t)(rbase + m) * H_;
  const bf16* arow1 = arow0 + (size_t)16 * H_;

  v8f acc[2][4];
#pragma unroll
  for (int r = 0; r < 2; ++r)
#pragma unroll
    for (int t = 0; t < 4; ++t) acc[r][t] = zero8();

  v16bf a0 = load_a_frag(arow0, hi);
  v16bf a1 = load_a_frag(arow1, hi);
  v16bf bfr[4];
#pragma unroll
  for (int t = 0; t < 4; ++t)
    bfr[t] = load_b_frag(Wt + (size_t)(n0 + t * 16 + m) * H_ + hi * 16);

#pragma unroll
  for (int kc = 0; kc < H_; kc += 32) {
    v16bf na0, na1, nb[4];
    if (kc + 32 < H_) {
      na0 = load_a_frag(arow0 + kc + 32, hi);
      na1 = load_a_frag(arow1 + kc + 32, hi);
#pragma unroll
      for (int t = 0; t < 4; ++t)
        nb[t] = load_b_frag(Wt + (size_t)(n0 + t * 16 + m) * H_ + kc + 32 + hi * 16);
    }
#pragma unroll
    for (int t = 0; t < 4; ++t) {
      acc[0][t] = wmma_bf16(a0, bfr[t], acc[0][t]);
      acc[1][t] = wmma_bf16(a1, bfr[t], acc[1][t]);
    }
    if (kc + 32 < H_) {
      a0 = na0; a1 = na1;
#pragma unroll
      for (int t = 0; t < 4; ++t) bfr[t] = nb[t];
    }
  }

#pragma unroll
  for (int r = 0; r < 2; ++r) {
#pragma unroll
    for (int t = 0; t < 4; ++t) {
      int cg = n0 + t * 16 + m;
      float bv = bias[cg];
#pragma unroll
      for (int e = 0; e < 8; ++e) {
        int row = rbase + r * 16 + e + 8 * hi;
        out[(size_t)row * H_ + cg] = acc[r][t][e] + bv;
      }
    }
  }
}

// ---------------- launch ----------------
extern "C" void kernel_launch(void* const* d_in, const int* in_sizes, int n_in,
                              void* d_out, int out_size, void* d_ws, size_t ws_size,
                              hipStream_t stream) {
  (void)in_sizes; (void)n_in; (void)out_size; (void)ws_size;
  const float* x    = (const float*)d_in[0];
  const float* geo  = (const float*)d_in[1];
  const float* Wqkv = (const float*)d_in[2];
  const float* bqkv = (const float*)d_in[3];
  const float* Wout = (const float*)d_in[4];
  const float* bout = (const float*)d_in[5];
  float* out = (float*)d_out;

  const size_t MN = (size_t)B_ * N_;   // 16384 rows
  char* ws = (char*)d_ws;
  size_t off = 0;
  auto alloc = [&](size_t bytes) {
    void* p = ws + off;
    off += (bytes + 255) & ~(size_t)255;
    return p;
  };
  bf16* Xb  = (bf16*)alloc(MN * H_ * sizeof(bf16));
  bf16* Qb  = (bf16*)alloc(MN * H_ * sizeof(bf16));
  bf16* Kb  = (bf16*)alloc(MN * H_ * sizeof(bf16));
  bf16* Vtg = (bf16*)alloc(MN * H_ * sizeof(bf16));  // per-batch [H][N]
  bf16* Ob  = (bf16*)alloc(MN * H_ * sizeof(bf16));
  bf16* WtQ = (bf16*)alloc((size_t)H_ * H3_ * sizeof(bf16));
  bf16* WtO = (bf16*)alloc((size_t)H_ * H_ * sizeof(bf16));

  int totX = (int)(MN * H_);
  k_f32_to_bf16<<<(totX + 255) / 256, 256, 0, stream>>>(x, Xb, totX);
  k_transpose_w<<<(H_ * H3_ + 255) / 256, 256, 0, stream>>>(Wqkv, WtQ, H_, H3_);
  k_transpose_w<<<(H_ * H_ + 255) / 256, 256, 0, stream>>>(Wout, WtO, H_, H_);

  dim3 g1((unsigned)(MN / 128), H3_ / 128);
  k_qkv_gemm<<<g1, 256, 0, stream>>>(Xb, WtQ, bqkv, Qb, Kb, Vtg);

  k_attn<<<B_ * (N_ / 64), 128, 0, stream>>>(Qb, Kb, Vtg, geo, Ob);

  dim3 g2((unsigned)(MN / 128), H_ / 128);
  k_out_gemm<<<g2, 256, 0, stream>>>(Ob, WtO, bout, out);
}
